// SinkhornNet_85143431676178
// MI455X (gfx1250) — compile-verified, matrix operand order checked by reading, not run on hardware
//
#include <hip/hip_runtime.h>
#include <hip/hip_bf16.h>
#include <math.h>

// ---------------------------------------------------------------------------
// Types for CDNA5 WMMA (wave32, 16x16x32 bf16 -> f32)
// ---------------------------------------------------------------------------
typedef __attribute__((ext_vector_type(16))) __bf16 v16bf;
typedef __attribute__((ext_vector_type(8)))  float  v8f;

union FragU { unsigned u[8]; v16bf v; };

__device__ __forceinline__ unsigned short f32_to_bf16(float f) {
  unsigned u = __builtin_bit_cast(unsigned, f);
  unsigned r = 0x7FFFu + ((u >> 16) & 1u);  // round-to-nearest-even
  return (unsigned short)((u + r) >> 16);
}

// Pack two f32 -> packed bf16x2 with one v_perm_b32 (round-half-up).
__device__ __forceinline__ unsigned permpack_bf16x2(float a, float b) {
  unsigned ua = __builtin_bit_cast(unsigned, a) + 0x8000u;
  unsigned ub = __builtin_bit_cast(unsigned, b) + 0x8000u;
  // dst = { ub[31:16], ua[31:16] } : byte sel 7,6 from src0(ub); 3,2 from src1(ua)
  return __builtin_amdgcn_perm(ub, ua, 0x07060302u);
}

// ---------------------------------------------------------------------------
// CDNA5 async copy Global->LDS (ASYNCcnt-tracked) + wait
// ---------------------------------------------------------------------------
__device__ __forceinline__ void async_g2l_b128(unsigned lds_addr, unsigned goff,
                                               const void* sbase) {
  asm volatile("global_load_async_to_lds_b128 %0, %1, %2"
               : : "v"(lds_addr), "v"(goff), "s"(sbase) : "memory");
}
template <int N>
__device__ __forceinline__ void wait_asynccnt() {
  asm volatile("s_wait_asynccnt %0" : : "n"(N) : "memory");
}

// ---------------------------------------------------------------------------
// B fragment straight from global: bf16 (N, Kpad) pre-transposed weights.
// ISA 16-bit 16x32 layout: lane L: n = L&15, h = L>>4;
// VGPR0-3 hold K = h*8 .. h*8+7, VGPR4-7 hold K = 16+h*8 .. 16+h*8+7.
// ---------------------------------------------------------------------------
__device__ __forceinline__ v16bf frag_from_bf16(const unsigned short* __restrict__ base,
                                                int ld, int kbase, int lane) {
  const int m = lane & 15, h = lane >> 4;
  const unsigned short* p = base + (size_t)m * ld + kbase + h * 8;
  const uint4 c0 = *reinterpret_cast<const uint4*>(p);
  const uint4 c1 = *reinterpret_cast<const uint4*>(p + 16);
  FragU f;
  f.u[0] = c0.x; f.u[1] = c0.y; f.u[2] = c0.z; f.u[3] = c0.w;
  f.u[4] = c1.x; f.u[5] = c1.y; f.u[6] = c1.z; f.u[7] = c1.w;
  return f.v;
}

// ---------------------------------------------------------------------------
// Weight prep: f32 (K, N) -> bf16 (N, Kpad) transposed, zero-padded K tail.
// ---------------------------------------------------------------------------
__global__ void prep_weight_kernel(const float* __restrict__ W, unsigned short* __restrict__ Wp,
                                   int K, int Nn, int Kpad) {
  int idx = blockIdx.x * blockDim.x + threadIdx.x;
  if (idx >= Nn * Kpad) return;
  int n = idx / Kpad;
  int k = idx - n * Kpad;
  unsigned short v = 0;
  if (k < K) v = f32_to_bf16(W[(size_t)k * Nn + n]);
  Wp[idx] = v;
}

// Copy x_pos (seq cols 2348..2351) into cat buffer cols 256..259; zero 260..287.
__global__ void pos_copy_kernel(const float* __restrict__ seq, unsigned short* __restrict__ catb) {
  int idx = blockIdx.x * blockDim.x + threadIdx.x;
  if (idx >= 32768 * 32) return;
  int m = idx >> 5, j = idx & 31;
  unsigned short v = 0;
  if (j < 4) v = f32_to_bf16(seq[(size_t)m * 2352 + 2348 + j]);
  catb[(size_t)m * 288 + 256 + j] = v;
}

// ---------------------------------------------------------------------------
// WMMA GEMM with LDS-staged A and async double-buffering.
//   Block = 128 threads (4 waves). Grid = (M/16, N/64).
//   All 4 waves share one 16-row A slab (staged by wave 0 via
//   GLOBAL_LOAD_ASYNC_TO_LDS_B128, ASYNCcnt double-buffered); each wave owns a
//   16-column N-tile and accumulates with v_wmma_f32_16x16x32_bf16.
//   A: row-major f32 (raw seq) or bf16 (intermediates), lda elems, col a_col0.
//   Bp: bf16 (N, Kpad) pre-transposed weights (zero-padded K tail, L2-resident).
//   Out: bf16 (+bias+relu) or f32 (+bias+tanh).
// ---------------------------------------------------------------------------
template <int A_IS_F32, int USE_TANH>
__global__ __launch_bounds__(128) void gemm_bf16_wmma(
    const void* __restrict__ A, int lda, int a_col0,
    const unsigned short* __restrict__ Bp, int kpad,
    const float* __restrict__ bias,
    void* __restrict__ Out, int ldo, int o_col0) {
  constexpr int ES = A_IS_F32 ? 4 : 2;     // element bytes in A
  constexpr int ROWB = 32 * ES;            // bytes per row of a 16x32 K-slice
  constexpr int TILEB = 16 * ROWB;         // slice bytes (2048 f32 / 1024 bf16)
  constexpr int NB = TILEB / 512;          // async b128 instructions per slice
  constexpr int CPR = ROWB / 16;           // 16B chunks per row

  __shared__ alignas(16) char atile[2][2048];

  const int tid = threadIdx.x;
  const int wave = tid >> 5;
  const int lane = tid & 31;
  const int m0 = blockIdx.x * 16;
  const int n0 = blockIdx.y * 64 + wave * 16;
  const int nk = kpad / 32;

  const char* Abase = (const char*)A + ((size_t)m0 * lda + a_col0) * ES;
  const unsigned short* Bt = Bp + (size_t)n0 * kpad;

  // wave 0: stage A K-slice kt into LDS buffer buf (NB async b128 per slice)
  auto issue_slice = [&](int kt, int buf) {
    const char* g = Abase + (size_t)kt * ROWB;           // uniform base
    unsigned lbase = (unsigned)(unsigned long long)(&atile[buf][0]);
#pragma unroll
    for (int i = 0; i < NB; ++i) {
      int chunk = i * 32 + lane;                          // 16B chunk id
      int row = chunk / CPR;
      int cb = (chunk - row * CPR) * 16;                  // byte within row
      unsigned goff = (unsigned)(row * lda * ES + cb);
      async_g2l_b128(lbase + (unsigned)chunk * 16u, goff, g);
    }
  };

  if (wave == 0) issue_slice(0, 0);

  v8f acc = {};
  const int m = lane & 15, h = lane >> 4;

  for (int kt = 0; kt < nk; ++kt) {
    if (wave == 0) {
      issue_slice((kt + 1) % nk, (kt + 1) & 1);  // prefetch next (wraps harmlessly)
      wait_asynccnt<NB>();                       // in-order: slice kt has landed
    }
    __syncthreads();  // slice kt visible to all 4 waves

    // A fragment from LDS (ds_load_b128), converting f32->bf16 if needed
    FragU fa;
    if (A_IS_F32) {
      const float* tp = (const float*)atile[kt & 1] + m * 32 + h * 8;
#pragma unroll
      for (int i = 0; i < 4; ++i) fa.u[i] = permpack_bf16x2(tp[2 * i], tp[2 * i + 1]);
#pragma unroll
      for (int i = 0; i < 4; ++i) fa.u[4 + i] = permpack_bf16x2(tp[16 + 2 * i], tp[16 + 2 * i + 1]);
    } else {
      const unsigned short* tp = (const unsigned short*)atile[kt & 1] + m * 32 + h * 8;
      const uint4 c0 = *reinterpret_cast<const uint4*>(tp);
      const uint4 c1 = *reinterpret_cast<const uint4*>(tp + 16);
      fa.u[0] = c0.x; fa.u[1] = c0.y; fa.u[2] = c0.z; fa.u[3] = c0.w;
      fa.u[4] = c1.x; fa.u[5] = c1.y; fa.u[6] = c1.z; fa.u[7] = c1.w;
    }

    v16bf b = frag_from_bf16(Bt, kpad, kt * 32, lane);
    acc = __builtin_amdgcn_wmma_f32_16x16x32_bf16(
        /*neg_a=*/false, fa.v, /*neg_b=*/false, b,
        /*c_mod=*/(short)0, acc, /*reuse_a=*/false, /*reuse_b=*/false);

    __syncthreads();  // all reads of this buffer done before it is overwritten
  }

  // C/D layout: VGPR r, lanes 0-15 -> M=r, lanes 16-31 -> M=8+r; N = lane&15.
  const int n = n0 + m;
  const float bv = bias[n];
#pragma unroll
  for (int r = 0; r < 8; ++r) {
    const int mm = m0 + r + 8 * h;
    float v = acc[r] + bv;
    if (USE_TANH) {
      ((float*)Out)[(size_t)mm * ldo + o_col0 + n] = tanhf(v);
    } else {
      v = v > 0.f ? v : 0.f;
      ((unsigned short*)Out)[(size_t)mm * ldo + o_col0 + n] = f32_to_bf16(v);
    }
  }
}

// ---------------------------------------------------------------------------
// Sinkhorn: one workgroup per batch element; 128x128 f32 matrix lives in LDS.
// x = exp(x/tau); 20x { col-normalize, row-normalize }. In-place on d_out.
// ---------------------------------------------------------------------------
__global__ __launch_bounds__(128) void sinkhorn_kernel(float* __restrict__ x) {
  __shared__ float s[128 * 128];  // 64 KB (LDS: 320 KB/WGP on CDNA5)
  const int b = blockIdx.x;
  const int t = threadIdx.x;  // 0..127
  float* base = x + (size_t)b * (128 * 128);

  for (int c = 0; c < 128; ++c)
    s[t * 128 + c] = __expf(base[t * 128 + c] * 10.0f);  // 1/TAU = 10
  __syncthreads();

  for (int it = 0; it < 20; ++it) {
    // axis=-2: thread t owns column t
    float cs = 0.f;
    for (int r = 0; r < 128; ++r) cs += s[r * 128 + t];
    float inv = 1.0f / (1e-7f + cs);
    for (int r = 0; r < 128; ++r) s[r * 128 + t] *= inv;
    __syncthreads();
    // axis=-1: thread t owns row t
    float rs = 0.f;
    for (int c = 0; c < 128; ++c) rs += s[t * 128 + c];
    inv = 1.0f / (1e-7f + rs);
    for (int c = 0; c < 128; ++c) s[t * 128 + c] *= inv;
    __syncthreads();
  }

  for (int c = 0; c < 128; ++c)
    base[t * 128 + c] = s[t * 128 + c];
}

// ---------------------------------------------------------------------------
// Host-side orchestration
// ---------------------------------------------------------------------------
extern "C" void kernel_launch(void* const* d_in, const int* in_sizes, int n_in,
                              void* d_out, int out_size, void* d_ws, size_t ws_size,
                              hipStream_t stream) {
  (void)in_sizes; (void)n_in; (void)out_size; (void)ws_size;

  const float* seq      = (const float*)d_in[0];
  const float* W1_txt   = (const float*)d_in[1];
  const float* b1_txt   = (const float*)d_in[2];
  const float* W1_vis   = (const float*)d_in[3];
  const float* b1_vis   = (const float*)d_in[4];
  const float* W2_vis   = (const float*)d_in[5];
  const float* b2_vis   = (const float*)d_in[6];
  const float* W_fc_pos = (const float*)d_in[7];
  const float* b_fc_pos = (const float*)d_in[8];
  const float* W_fc     = (const float*)d_in[9];
  const float* b_fc     = (const float*)d_in[10];
  float* out = (float*)d_out;

  // Workspace carve-up (256B aligned): prepped bf16 weights + bf16 activations.
  char* ws = (char*)d_ws;
  size_t off = 0;
  auto take = [&](size_t bytes) -> char* {
    char* p = ws + off;
    off += (bytes + 255) & ~(size_t)255;
    return p;
  };
  unsigned short* W1_txt_p   = (unsigned short*)take((size_t)128 * 320 * 2);
  unsigned short* W1_vis_p   = (unsigned short*)take((size_t)512 * 2048 * 2);
  unsigned short* W2_vis_p   = (unsigned short*)take((size_t)128 * 512 * 2);
  unsigned short* W_fc_pos_p = (unsigned short*)take((size_t)256 * 288 * 2);
  unsigned short* W_fc_p     = (unsigned short*)take((size_t)128 * 256 * 2);
  unsigned short* visb       = (unsigned short*)take((size_t)32768 * 512 * 2);
  unsigned short* catb       = (unsigned short*)take((size_t)32768 * 288 * 2);
  unsigned short* fcb        = (unsigned short*)take((size_t)32768 * 256 * 2);

  auto cdiv = [](int a, int b) { return (a + b - 1) / b; };

  // Weight conversion + transpose (+ K zero-padding)
  prep_weight_kernel<<<cdiv(128 * 320, 256), 256, 0, stream>>>(W1_txt,   W1_txt_p,   300,  128, 320);
  prep_weight_kernel<<<cdiv(512 * 2048, 256), 256, 0, stream>>>(W1_vis,  W1_vis_p,   2048, 512, 2048);
  prep_weight_kernel<<<cdiv(128 * 512, 256), 256, 0, stream>>>(W2_vis,   W2_vis_p,   512,  128, 512);
  prep_weight_kernel<<<cdiv(256 * 288, 256), 256, 0, stream>>>(W_fc_pos, W_fc_pos_p, 260,  256, 288);
  prep_weight_kernel<<<cdiv(128 * 256, 256), 256, 0, stream>>>(W_fc,     W_fc_p,     256,  128, 256);

  // x_pos -> cat cols 256..259 (pad 260..287 zero)
  pos_copy_kernel<<<cdiv(32768 * 32, 256), 256, 0, stream>>>(seq, catb);

  const int MT = 32768 / 16;  // 2048 M-tiles; block covers 64 N with 4 waves

  // 1) x_txt = relu(seq[:, :300] @ W1_txt + b1_txt)  -> cat[:, 0:128]
  gemm_bf16_wmma<1, 0><<<dim3(MT, 2), 128, 0, stream>>>(seq, 2352, 0,    W1_txt_p,   320,  b1_txt,   catb, 288, 0);
  // 2) v1 = relu(seq[:, 300:2348] @ W1_vis + b1_vis) -> visb (32768 x 512)
  gemm_bf16_wmma<1, 0><<<dim3(MT, 8), 128, 0, stream>>>(seq, 2352, 300, W1_vis_p,   2048, b1_vis,   visb, 512, 0);
  // 3) v2 = relu(v1 @ W2_vis + b2_vis)               -> cat[:, 128:256]
  gemm_bf16_wmma<0, 0><<<dim3(MT, 2), 128, 0, stream>>>(visb, 512, 0,    W2_vis_p,   512,  b2_vis,   catb, 288, 128);
  // 4) h = relu(cat @ W_fc_pos + b_fc_pos)           -> fcb (32768 x 256)
  gemm_bf16_wmma<0, 0><<<dim3(MT, 4), 128, 0, stream>>>(catb, 288, 0,   W_fc_pos_p, 288,  b_fc_pos, fcb,  256, 0);
  // 5) y = tanh(h @ W_fc + b_fc)                     -> d_out f32 (32768 x 128)
  gemm_bf16_wmma<0, 1><<<dim3(MT, 2), 128, 0, stream>>>(fcb, 256, 0,     W_fc_p,     256,  b_fc,     out,  128, 0);

  // Sinkhorn in LDS, in-place on d_out
  sinkhorn_kernel<<<256, 128, 0, stream>>>(out);
}
